// SASRec_14929306321627
// MI455X (gfx1250) — compile-verified, matrix operand order checked by reading, not run on hardware
//
#include <hip/hip_runtime.h>
#include <hip/hip_bf16.h>
#include <math.h>

// Problem constants (match reference)
#define BD   4
#define SD   2048
#define HD   512
#define NHD  8
#define DHD  64
#define IDIM 2048
#define LDL  2
#define MD   (BD * SD)   // 8192 rows of activations

typedef unsigned short u16;
typedef __attribute__((ext_vector_type(16))) __bf16 bf16x16;
typedef __attribute__((ext_vector_type(8)))  float  f32x8;
typedef __attribute__((ext_vector_type(4)))  unsigned int u32x4;
typedef __attribute__((ext_vector_type(8)))  int  i32x8;
typedef __attribute__((ext_vector_type(4)))  int  i32x4;

__device__ __forceinline__ u16 f2bf(float f) {
  unsigned u = __float_as_uint(f);
  u += 0x7fffu + ((u >> 16) & 1u);   // round-to-nearest-even
  return (u16)(u >> 16);
}

// Load a 16-element bf16 WMMA fragment: elements [p..p+7] -> VGPR0-3,
// [p+16..p+23] -> VGPR4-7 (ISA 16-bit A/B layout; caller picks p per lane).
__device__ __forceinline__ bf16x16 ld_frag(const u16* p) {
  union { bf16x16 v; uint4 q[2]; } u;
  u.q[0] = *reinterpret_cast<const uint4*>(p);
  u.q[1] = *reinterpret_cast<const uint4*>(p + 16);
  return u.v;
}

// Generic LDS pointer -> raw LDS byte offset (aperture: addr[31:0] is the
// LDS address; upper bits are the shared aperture base).
__device__ __forceinline__ unsigned lds_off(const void* p) {
  return (unsigned)(unsigned long long)(size_t)p;
}

// ---------------------------------------------------------------------------
// Tensor Data Mover: async 2-D tile (global -> LDS), bf16 elements, with
// hardware LDS padding. D# per CDNA5 ISA ch.8 (group0: count/lds/global/type;
// group1: data_size, pad ctrl, dims, dim0 stride). 2-D => groups 2/3 zero.
//   tile_w    : tile dim0 in elements (row length)
//   tile_h    : tile dim1 (rows)
//   stride    : global row stride in elements
//   pad_int   : pad_interval code (3 => every 16 DWORDs, 4 => every 32 DWORDs)
//   pad_amt   : pad_amount  code (3 => 4 DWORDs = 16 bytes)
// ---------------------------------------------------------------------------
__device__ __forceinline__ void tdm_load_2d(unsigned lds_addr,
                                            unsigned long long gaddr,
                                            unsigned tile_w, unsigned tile_h,
                                            unsigned stride,
                                            unsigned pad_int, unsigned pad_amt) {
  u32x4 g0;
  g0[0] = 1u;                                            // count=1 (valid user D#)
  g0[1] = lds_addr;                                      // lds_addr
  g0[2] = (unsigned)(gaddr & 0xffffffffull);             // global_addr[31:0]
  g0[3] = (unsigned)((gaddr >> 32) & 0x1ffffffull)       // global_addr[56:32]
        | (2u << 30);                                    // type=2 ("image")
  const unsigned tdim1 = 1u << 20;                       // large (no OOB clip)
  i32x8 g1;
  g1[0] = (int)((1u << 16)                               // data_size=1 -> 2 bytes
        | (1u << 20)                                     // pad_enable
        | (pad_int << 22) | (pad_amt << 25));
  g1[1] = (int)((stride & 0xffffu) << 16);               // tensor_dim0[15:0] = stride
  g1[2] = (int)((stride >> 16) | ((tdim1 & 0xffffu) << 16)); // dim0 hi | dim1 lo
  g1[3] = (int)((tdim1 >> 16) | (tile_w << 16));         // dim1 hi | tile_dim0
  g1[4] = (int)(tile_h & 0xffffu);                       // tile_dim1 (tile_dim2=0)
  g1[5] = (int)stride;                                   // tensor_dim0_stride[31:0]
  g1[6] = 0;                                             // stride hi | dim1_stride lo
  g1[7] = 0;
  i32x4 gz4 = {0, 0, 0, 0};
  i32x8 gz8 = {0, 0, 0, 0, 0, 0, 0, 0};
  __builtin_amdgcn_tensor_load_to_lds(g0, g1, gz4, gz4, gz8, 0);
}

// ---------------------------------------------------------------------------
// Elementwise converts
// ---------------------------------------------------------------------------
__global__ void init_h_kernel(const float* __restrict__ x, float* __restrict__ h,
                              u16* __restrict__ hbf, int n) {
  int i = blockIdx.x * 256 + threadIdx.x;
  if (i < n) { float v = x[i]; h[i] = v; hbf[i] = f2bf(v); }
}

// src: K x N fp32 (row-major)  ->  dst: N x K bf16 (row-major)   (W^T)
__global__ void conv_wt_kernel(const float* __restrict__ src, u16* __restrict__ dst,
                               int K, int N) {
  int i = blockIdx.x * 256 + threadIdx.x;
  if (i < N * K) {
    int n = i / K, k = i - n * K;
    dst[i] = f2bf(src[(size_t)k * N + n]);
  }
}

// ---------------------------------------------------------------------------
// bf16 WMMA GEMM:  C[M,N] = A[M,K] * Bt[N,K]^T + bias, optional GELU epilogue.
// Block: 256 thr (8 waves), tile 128x128, K-step 32. Wave owns 32x64.
// Tiles staged by the Tensor Data Mover (wave 0 issues, TENSORcnt-tracked),
// double-buffered so the DMA of step k+1 overlaps WMMA of step k.
// ---------------------------------------------------------------------------
template<int EPI, bool OUTBF>
__global__ __launch_bounds__(256) void gemm_bf16_kernel(
    const u16* __restrict__ A, const u16* __restrict__ Bt,
    const float* __restrict__ bias, void* __restrict__ Cout,
    int M, int N, int K)
{
  constexpr int BM = 128, BN = 128, BK = 32, LDT = 40;  // 80B pitch (TDM pad)
  __shared__ u16 sA[2][BM * LDT];
  __shared__ u16 sB[2][BN * LDT];

  const int tid  = threadIdx.x;
  const int lane = tid & 31;
  const int wave = tid >> 5;
  const int wm   = wave >> 1;          // 0..3 -> 32-row slab
  const int wn   = wave & 1;           // 0..1 -> 64-col slab
  const int m0   = blockIdx.y * BM;
  const int n0   = blockIdx.x * BN;
  const int half = lane >> 4;
  const int l16  = lane & 15;
  const int koff = half ? 8 : 0;

  const unsigned long long gA = (unsigned long long)(size_t)A;
  const unsigned long long gB = (unsigned long long)(size_t)Bt;

  // Prime the pipeline: DMA tile 0 into buffer 0.
  if (wave == 0) {
    tdm_load_2d(lds_off(&sA[0][0]), gA + ((unsigned long long)m0 * K) * 2,
                BK, BM, (unsigned)K, 3, 3);
    tdm_load_2d(lds_off(&sB[0][0]), gB + ((unsigned long long)n0 * K) * 2,
                BK, BN, (unsigned)K, 3, 3);
  }

  f32x8 acc[2][4] = {};

  const int nks = K / BK;
  for (int ks = 0; ks < nks; ++ks) {
    const int buf = ks & 1;
    if (wave == 0) __builtin_amdgcn_s_wait_tensorcnt(0);  // tile[buf] landed
    __syncthreads();   // tile[buf] visible; everyone done reading tile[buf^1]
    if (wave == 0 && ks + 1 < nks) {
      const int k1 = (ks + 1) * BK;
      tdm_load_2d(lds_off(&sA[buf ^ 1][0]),
                  gA + ((unsigned long long)m0 * K + k1) * 2, BK, BM, (unsigned)K, 3, 3);
      tdm_load_2d(lds_off(&sB[buf ^ 1][0]),
                  gB + ((unsigned long long)n0 * K + k1) * 2, BK, BN, (unsigned)K, 3, 3);
    }

    bf16x16 af0 = ld_frag(&sA[buf][(wm * 32 +  0 + l16) * LDT + koff]);
    bf16x16 af1 = ld_frag(&sA[buf][(wm * 32 + 16 + l16) * LDT + koff]);
#pragma unroll
    for (int nt = 0; nt < 4; ++nt) {
      bf16x16 bf = ld_frag(&sB[buf][(wn * 64 + nt * 16 + l16) * LDT + koff]);
      acc[0][nt] = __builtin_amdgcn_wmma_f32_16x16x32_bf16(
          false, af0, false, bf, (short)0, acc[0][nt], false, false);
      acc[1][nt] = __builtin_amdgcn_wmma_f32_16x16x32_bf16(
          false, af1, false, bf, (short)0, acc[1][nt], false, false);
    }
  }

#pragma unroll
  for (int mt = 0; mt < 2; ++mt) {
#pragma unroll
    for (int nt = 0; nt < 4; ++nt) {
      const int col = n0 + wn * 64 + nt * 16 + l16;
      const float bv = bias[col];
#pragma unroll
      for (int r = 0; r < 8; ++r) {
        const int row = m0 + wm * 32 + mt * 16 + r + half * 8;
        float v = acc[mt][nt][r] + bv;
        if (EPI == 1) v = 0.5f * v * (1.0f + erff(v * 0.70710678118654752f));
        if (OUTBF) ((u16*)Cout)[(size_t)row * N + col] = f2bf(v);
        else       ((float*)Cout)[(size_t)row * N + col] = v;
      }
    }
  }
}

// ---------------------------------------------------------------------------
// Flash attention (causal), bf16 in/out, fp32 online softmax.
// Grid: (S/64, NH, B). Block: 128 thr (4 waves); wave owns 16 query rows.
// K-tile staged by TDM; V transposed manually (TDM cannot transpose).
// ---------------------------------------------------------------------------
__global__ __launch_bounds__(128) void flash_attn_kernel(
    const u16* __restrict__ Q, const u16* __restrict__ K,
    const u16* __restrict__ V, u16* __restrict__ O)
{
  constexpr int LKT = 72;  // K-tile pitch (64+8): 144B (TDM pad 4 DW / 32 DW)
  constexpr int LVT = 40;  // Vt / P pitch (32+8)
  __shared__ u16 sK [32 * LKT];
  __shared__ u16 sVt[64 * LVT];
  __shared__ u16 sP [4 * 16 * LVT];

  const int tid  = threadIdx.x;
  const int lane = tid & 31;
  const int wave = tid >> 5;
  const int half = lane >> 4;
  const int l16  = lane & 15;
  const int koff = half ? 8 : 0;
  const int qb   = blockIdx.x;
  const int head = blockIdx.y;
  const int b    = blockIdx.z;
  const float scale = 0.125f;  // 1/sqrt(64)
  const size_t hoff = (size_t)head * DHD;

  // Q fragments for this wave's 16 rows, held for the whole loop
  const int qrow_l = qb * 64 + wave * 16 + l16;
  bf16x16 aQ0 = ld_frag(&Q[(size_t)(b * SD + qrow_l) * HD + hoff +  0 + koff]);
  bf16x16 aQ1 = ld_frag(&Q[(size_t)(b * SD + qrow_l) * HD + hoff + 32 + koff]);

  f32x8 acc[4] = {};
  float m_i[8], l_i[8];
#pragma unroll
  for (int r = 0; r < 8; ++r) { m_i[r] = -1e30f; l_i[r] = 0.0f; }

  const int ldr = tid >> 2;           // 0..31: key row loaded by this thread
  const int ldc = (tid & 3) * 16;     // 0/16/32/48: dh offset
  const int nkt = qb * 2 + 2;         // causal: keys only up to this q-block
  const unsigned long long gK = (unsigned long long)(size_t)K;

  for (int kt = 0; kt < nkt; ++kt) {
    const int key0 = kt * 32;
    __syncthreads();  // previous iteration's LDS consumers done
    // K tile 32x64 via Tensor Data Mover (row stride H, pad to 72 u16)
    if (wave == 0) {
      tdm_load_2d(lds_off(&sK[0]),
                  gK + ((unsigned long long)(b * SD + key0) * HD + hoff) * 2,
                  64, 32, (unsigned)HD, 4, 3);
    }
    {
      // V staged transposed: sVt[dh][key]
      const u16* gv = V + (size_t)(b * SD + key0 + ldr) * HD + hoff + ldc;
      union { uint4 q[2]; u16 s[16]; } uv;
      uv.q[0] = *reinterpret_cast<const uint4*>(gv);
      uv.q[1] = *reinterpret_cast<const uint4*>(gv + 8);
#pragma unroll
      for (int i = 0; i < 16; ++i)
        sVt[(ldc + i) * LVT + ldr] = uv.s[i];
    }
    if (wave == 0) __builtin_amdgcn_s_wait_tensorcnt(0);
    __syncthreads();

    // S = Q * K^T : 16x32 scores per wave (two 16x16 tiles, two K-steps)
    f32x8 s0 = {}, s1 = {};
    {
      bf16x16 bk00 = ld_frag(&sK[( 0 + l16) * LKT +  0 + koff]);
      bf16x16 bk10 = ld_frag(&sK[(16 + l16) * LKT +  0 + koff]);
      s0 = __builtin_amdgcn_wmma_f32_16x16x32_bf16(false, aQ0, false, bk00, (short)0, s0, false, false);
      s1 = __builtin_amdgcn_wmma_f32_16x16x32_bf16(false, aQ0, false, bk10, (short)0, s1, false, false);
      bf16x16 bk01 = ld_frag(&sK[( 0 + l16) * LKT + 32 + koff]);
      bf16x16 bk11 = ld_frag(&sK[(16 + l16) * LKT + 32 + koff]);
      s0 = __builtin_amdgcn_wmma_f32_16x16x32_bf16(false, aQ1, false, bk01, (short)0, s0, false, false);
      s1 = __builtin_amdgcn_wmma_f32_16x16x32_bf16(false, aQ1, false, bk11, (short)0, s1, false, false);
    }

    // Online softmax; rows live across 16-lane halves in the C layout
    float p0[8], p1[8];
#pragma unroll
    for (int r = 0; r < 8; ++r) {
      const int qrow = qb * 64 + wave * 16 + r + half * 8;
      float x0 = s0[r] * scale + ((key0      + l16 > qrow) ? -1e4f : 0.0f);
      float x1 = s1[r] * scale + ((key0 + 16 + l16 > qrow) ? -1e4f : 0.0f);
      float mx = fmaxf(x0, x1);
#pragma unroll
      for (int o = 8; o >= 1; o >>= 1) mx = fmaxf(mx, __shfl_xor(mx, o, 32));
      const float mnew = fmaxf(m_i[r], mx);
      const float corr = __expf(m_i[r] - mnew);
      m_i[r] = mnew;
      const float e0 = __expf(x0 - mnew);
      const float e1 = __expf(x1 - mnew);
      float rs = e0 + e1;
#pragma unroll
      for (int o = 8; o >= 1; o >>= 1) rs += __shfl_xor(rs, o, 32);
      l_i[r] = l_i[r] * corr + rs;
#pragma unroll
      for (int nt = 0; nt < 4; ++nt) acc[nt][r] *= corr;
      p0[r] = e0; p1[r] = e1;
    }

    // P: C-layout -> A-layout via wave-private LDS region
    {
      u16* wpP = &sP[wave * 16 * LVT];
#pragma unroll
      for (int r = 0; r < 8; ++r) {
        wpP[(r + half * 8) * LVT + l16]      = f2bf(p0[r]);
        wpP[(r + half * 8) * LVT + 16 + l16] = f2bf(p1[r]);
      }
    }
    __syncthreads();

    bf16x16 aP = ld_frag(&sP[(wave * 16 + l16) * LVT + koff]);
#pragma unroll
    for (int nt = 0; nt < 4; ++nt) {
      bf16x16 bv = ld_frag(&sVt[(nt * 16 + l16) * LVT + koff]);
      acc[nt] = __builtin_amdgcn_wmma_f32_16x16x32_bf16(
          false, aP, false, bv, (short)0, acc[nt], false, false);
    }
  }

#pragma unroll
  for (int nt = 0; nt < 4; ++nt) {
#pragma unroll
    for (int r = 0; r < 8; ++r) {
      const int qrow = qb * 64 + wave * 16 + r + half * 8;
      O[(size_t)(b * SD + qrow) * HD + hoff + nt * 16 + l16] = f2bf(acc[nt][r] / l_i[r]);
    }
  }
}

// ---------------------------------------------------------------------------
// h = LayerNorm(src)*g + b + h  (per row of H=512); also emits bf16 copy.
// One wave per row; 16 elements per lane.
// ---------------------------------------------------------------------------
__global__ __launch_bounds__(32) void ln_residual_kernel(
    const float* __restrict__ src, float* __restrict__ h, u16* __restrict__ hbf,
    const float* __restrict__ g, const float* __restrict__ bta)
{
  const int row  = blockIdx.x;
  const int lane = threadIdx.x;
  const size_t base = (size_t)row * HD;

  float xv[16];
  float s = 0.0f;
#pragma unroll
  for (int j = 0; j < 16; ++j) { xv[j] = src[base + j * 32 + lane]; s += xv[j]; }
#pragma unroll
  for (int o = 16; o >= 1; o >>= 1) s += __shfl_xor(s, o, 32);
  const float mean = s * (1.0f / HD);

  float vs = 0.0f;
#pragma unroll
  for (int j = 0; j < 16; ++j) { float d = xv[j] - mean; vs += d * d; }
#pragma unroll
  for (int o = 16; o >= 1; o >>= 1) vs += __shfl_xor(vs, o, 32);
  const float rstd = rsqrtf(vs * (1.0f / HD) + 1e-12f);

#pragma unroll
  for (int j = 0; j < 16; ++j) {
    const int idx = j * 32 + lane;
    const float y = (xv[j] - mean) * rstd * g[idx] + bta[idx] + h[base + idx];
    h[base + idx] = y;
    hbf[base + idx] = f2bf(y);
  }
}

// ---------------------------------------------------------------------------
// Host-side orchestration
// ---------------------------------------------------------------------------
extern "C" void kernel_launch(void* const* d_in, const int* in_sizes, int n_in,
                              void* d_out, int out_size, void* d_ws, size_t ws_size,
                              hipStream_t stream) {
  (void)in_sizes; (void)n_in; (void)out_size; (void)ws_size;

  const float* x   = (const float*)d_in[0];
  const float* Wq  = (const float*)d_in[2];
  const float* bq  = (const float*)d_in[3];
  const float* Wk  = (const float*)d_in[4];
  const float* bk  = (const float*)d_in[5];
  const float* Wv  = (const float*)d_in[6];
  const float* bv  = (const float*)d_in[7];
  const float* Wo  = (const float*)d_in[8];
  const float* bo  = (const float*)d_in[9];
  const float* g1  = (const float*)d_in[10];
  const float* be1 = (const float*)d_in[11];
  const float* W1  = (const float*)d_in[12];
  const float* b1  = (const float*)d_in[13];
  const float* W2  = (const float*)d_in[14];
  const float* b2  = (const float*)d_in[15];
  const float* g2  = (const float*)d_in[16];
  const float* be2 = (const float*)d_in[17];

  char* wp = (char*)d_ws;
  auto carve = [&](size_t bytes) -> void* {
    void* r = (void*)wp;
    wp += (bytes + 255) & ~(size_t)255;
    return r;
  };

  float* h    = (float*)carve((size_t)MD * HD * 4);
  u16*   hbf  = (u16*)  carve((size_t)MD * HD * 2);
  u16*   qb_  = (u16*)  carve((size_t)MD * HD * 2);
  u16*   kb_  = (u16*)  carve((size_t)MD * HD * 2);
  u16*   vb_  = (u16*)  carve((size_t)MD * HD * 2);
  u16*   ctxb = (u16*)  carve((size_t)MD * HD * 2);
  float* tmpf = (float*)carve((size_t)MD * HD * 4);
  u16*   ubf  = (u16*)  carve((size_t)MD * IDIM * 2);
  u16*   Wqt  = (u16*)  carve((size_t)LDL * HD * HD * 2);
  u16*   Wkt  = (u16*)  carve((size_t)LDL * HD * HD * 2);
  u16*   Wvt  = (u16*)  carve((size_t)LDL * HD * HD * 2);
  u16*   Wot  = (u16*)  carve((size_t)LDL * HD * HD * 2);
  u16*   W1t  = (u16*)  carve((size_t)LDL * HD * IDIM * 2);
  u16*   W2t  = (u16*)  carve((size_t)LDL * IDIM * HD * 2);

  const int nAct = MD * HD;
  init_h_kernel<<<(nAct + 255) / 256, 256, 0, stream>>>(x, h, hbf, nAct);

  // Pre-transpose + bf16-convert all weights
  for (int l = 0; l < LDL; ++l) {
    const int hh = HD * HD, hi = HD * IDIM;
    conv_wt_kernel<<<(hh + 255) / 256, 256, 0, stream>>>(Wq + (size_t)l * hh, Wqt + (size_t)l * hh, HD, HD);
    conv_wt_kernel<<<(hh + 255) / 256, 256, 0, stream>>>(Wk + (size_t)l * hh, Wkt + (size_t)l * hh, HD, HD);
    conv_wt_kernel<<<(hh + 255) / 256, 256, 0, stream>>>(Wv + (size_t)l * hh, Wvt + (size_t)l * hh, HD, HD);
    conv_wt_kernel<<<(hh + 255) / 256, 256, 0, stream>>>(Wo + (size_t)l * hh, Wot + (size_t)l * hh, HD, HD);
    conv_wt_kernel<<<(hi + 255) / 256, 256, 0, stream>>>(W1 + (size_t)l * hi, W1t + (size_t)l * hi, HD, IDIM);
    conv_wt_kernel<<<(hi + 255) / 256, 256, 0, stream>>>(W2 + (size_t)l * hi, W2t + (size_t)l * hi, IDIM, HD);
  }

  const dim3 gemmBlk(256);
  const dim3 gemmH(HD / 128, MD / 128);     // N=512
  const dim3 gemmI(IDIM / 128, MD / 128);   // N=2048
  const dim3 faGrid(SD / 64, NHD, BD);

  for (int l = 0; l < LDL; ++l) {
    const size_t hh = (size_t)l * HD * HD, hi = (size_t)l * HD * IDIM;
    const size_t ho = (size_t)l * HD, io = (size_t)l * IDIM;

    // QKV projections (bf16 out)
    gemm_bf16_kernel<0, true><<<gemmH, gemmBlk, 0, stream>>>(hbf, Wqt + hh, bq + ho, qb_, MD, HD, HD);
    gemm_bf16_kernel<0, true><<<gemmH, gemmBlk, 0, stream>>>(hbf, Wkt + hh, bk + ho, kb_, MD, HD, HD);
    gemm_bf16_kernel<0, true><<<gemmH, gemmBlk, 0, stream>>>(hbf, Wvt + hh, bv + ho, vb_, MD, HD, HD);

    // Causal flash attention
    flash_attn_kernel<<<faGrid, dim3(128), 0, stream>>>(qb_, kb_, vb_, ctxb);

    // Output projection (fp32 out) -> LN1 + residual
    gemm_bf16_kernel<0, false><<<gemmH, gemmBlk, 0, stream>>>(ctxb, Wot + hh, bo + ho, tmpf, MD, HD, HD);
    ln_residual_kernel<<<MD, 32, 0, stream>>>(tmpf, h, hbf, g1 + ho, be1 + ho);

    // FFN: GELU(h W1 + b1) W2 + b2 -> LN2 + residual
    gemm_bf16_kernel<1, true><<<gemmI, gemmBlk, 0, stream>>>(hbf, W1t + hi, b1 + io, ubf, MD, IDIM, HD);
    gemm_bf16_kernel<0, false><<<gemmH, gemmBlk, 0, stream>>>(ubf, W2t + hi, b2 + ho, tmpf, MD, HD, IDIM);
    ln_residual_kernel<<<MD, 32, 0, stream>>>(tmpf, h, hbf, g2 + ho, be2 + ho);
  }

  (void)hipMemcpyAsync(d_out, h, (size_t)MD * HD * 4, hipMemcpyDeviceToDevice, stream);
}
